// PositionPredictor_66657892434507
// MI455X (gfx1250) — matrix operand order repaired and checked by reading, not compiled, not run on hardware
//
#include <hip/hip_runtime.h>

// ---------------------------------------------------------------------------
// 3-layer KAN (efficient-KAN style) for MI455X / gfx1250, wave32 + WMMA.
//
// Each layer  y = silu(x) @ Wb^T + sum_{i,g} B_ig(x_i) * Ws[o,i,g]
// is expressed as a dense GEMM with K = 16 * in_features:
//   per-feature K-slot layout (16 halves): [silu(x_f), b0..b12, pad, pad]
// and a matching column-major f16 weight image in LDS:
//   col o, slot f*16+j : j==0 -> base_w[o,f]; 1..13 -> spline_w[o,f,j-1]; else 0
//
// Basis trick: the grid is uniform & extended, so every basis is a shifted
// cardinal cubic B-spline. For u = 5x+8 in [0,16), i=floor(u), w=u-i:
//   b_{i-3..i} = { (1-w)^3, 3w^3-6w^2+4, -3w^3+3w^2+3w+1, w^3 } / 6
// Scatter is fully branchless: saturating v_cvt_i32_f32 makes u>=16
// self-redirect via the unsigned range checks, u<0 is fixed with one cndmask,
// and any invalid index is steered to padding slot 15 (zero weights there).
// ---------------------------------------------------------------------------

#define WAVES 4
#define TPB   (WAVES * 32)
#define ROWS_PER_BLOCK (WAVES * 32)

typedef _Float16 v16h __attribute__((ext_vector_type(16)));
typedef _Float16 v8h  __attribute__((ext_vector_type(8)));
typedef float    v8f  __attribute__((ext_vector_type(8)));

__device__ __forceinline__ v8f wmma16(v16h a, v16h b, v8f c) {
    // D = A(16x32 f16) * B(32x16 f16) + C(16x16 f32)
    return __builtin_amdgcn_wmma_f32_16x16x32_f16(
        /*neg_a=*/false, a, /*neg_b=*/false, b,
        /*c_mod=*/(short)0, c, /*reuse_a=*/false, /*reuse_b=*/false);
}

// Build a v16h fragment from two 16-byte LDS pieces.
__device__ __forceinline__ v16h frag16(const _Float16* p_lo, const _Float16* p_hi) {
    v8h lo = *(const v8h*)p_lo;
    v8h hi = *(const v8h*)p_hi;
    return __builtin_shufflevector(lo, hi,
        0, 1, 2, 3, 4, 5, 6, 7, 8, 9, 10, 11, 12, 13, 14, 15);
}

// Emit one feature's 16 K-slot activations into LDS: [silu(v), b0..b12, 0, *]
__device__ __forceinline__ void kan_feature(_Float16* slot, float v) {
    // silu
    float s = v * __builtin_amdgcn_rcpf(1.0f + __expf(-v));
    // cardinal cubic B-spline, uniform extended grid on [-1,1], h = 0.2
    float u = __fmaf_rn(v, 5.0f, 8.0f);
    int   i = (int)u;                 // saturating cvt: u>=16 self-redirects
    i = (u < 0.0f) ? 0x40000 : i;     // negative u -> force out-of-range

    // zero the slot; silu rides element 0 of the first 16-byte store
    v8h z0 = {};
    z0[0] = (_Float16)s;
    v8h z1 = {};
    *(v8h*)(slot)     = z0;
    *(v8h*)(slot + 8) = z1;

    float w  = u - (float)i;          // garbage when redirected (harmless)
    float om = 1.0f - w;
    float w2 = w * w, w3 = w2 * w;
    const float k6 = 1.0f / 6.0f;
    float n0 = om * om * om * k6;
    float n1 = (3.0f * w3 - 6.0f * w2 + 4.0f) * k6;
    float n2 = (-3.0f * w3 + 3.0f * w2 + 3.0f * w + 1.0f) * k6;
    float n3 = w3 * k6;

    // basis index t = i-3+m valid iff t in [0,12]; slot index = t+1 else 15
    int i0 = ((unsigned)(i - 3) <= 12u) ? (i - 2) : 15;
    int i1 = ((unsigned)(i - 2) <= 12u) ? (i - 1) : 15;
    int i2 = ((unsigned)(i - 1) <= 12u) ? (i    ) : 15;
    int i3 = ((unsigned)(i    ) <= 12u) ? (i + 1) : 15;
    slot[i0] = (_Float16)n0;
    slot[i1] = (_Float16)n1;
    slot[i2] = (_Float16)n2;
    slot[i3] = (_Float16)n3;
}

#define WAVE_DS_FENCE() asm volatile("s_wait_dscnt 0x0" ::: "memory")

__global__ __launch_bounds__(TPB) void kan3_wmma_kernel(
    const float* __restrict__ X,
    const float* __restrict__ bw1, const float* __restrict__ sw1,
    const float* __restrict__ bw2, const float* __restrict__ sw2,
    const float* __restrict__ bw3, const float* __restrict__ sw3,
    float* __restrict__ Out, int nTiles)
{
    // Column-major f16 weight images (col stride = padded K)
    __shared__ __align__(16) _Float16 W1[32 * 64];   //  4 KB  (K=48 pad 64,  N=32)
    __shared__ __align__(16) _Float16 W2[16 * 512];  // 16 KB  (K=512,       N=16)
    __shared__ __align__(16) _Float16 W3[16 * 256];  //  8 KB  (K=256,       N=16, col0 real)
    __shared__ __align__(16) _Float16 Abuf[WAVES][32][32]; // 8 KB  per-wave A staging
    __shared__ __align__(16) float    Hbuf[WAVES][32][32]; // 16 KB per-wave D transpose

    const int tid = threadIdx.x;

    // ---- one-time weight image build (f32 -> f16, padded layout) ----
    for (int idx = tid; idx < 32 * 64; idx += TPB) {
        int o = idx >> 6, k = idx & 63, f = k >> 4, j = k & 15;
        float v = 0.0f;
        if (f < 3) {
            if (j == 0)       v = bw1[o * 3 + f];
            else if (j <= 13) v = sw1[(o * 3 + f) * 13 + (j - 1)];
        }
        W1[o * 64 + k] = (_Float16)v;
    }
    for (int idx = tid; idx < 16 * 512; idx += TPB) {
        int o = idx >> 9, k = idx & 511, f = k >> 4, j = k & 15;
        float v = 0.0f;
        if (j == 0)       v = bw2[o * 32 + f];
        else if (j <= 13) v = sw2[(o * 32 + f) * 13 + (j - 1)];
        W2[o * 512 + k] = (_Float16)v;
    }
    for (int idx = tid; idx < 16 * 256; idx += TPB) {
        int o = idx >> 8, k = idx & 255, f = k >> 4, j = k & 15;
        float v = 0.0f;
        if (o == 0) {
            if (j == 0)       v = bw3[f];
            else if (j <= 13) v = sw3[f * 13 + (j - 1)];
        }
        W3[o * 256 + k] = (_Float16)v;
    }
    __syncthreads();

    const int wv    = tid >> 5;
    const int lane  = tid & 31;
    const int l15   = lane & 15;
    const int hi    = lane >> 4;        // half-wave select
    const int hsB   = hi ? 16 : 0;      // B-frag: K offset (halves)
    const int hsA   = hi ? 8  : 0;      // A-frag: K offset within 16-half piece
    const int mhi   = hi ? 8  : 0;      // D layout: row offset per half-wave

    _Float16*       slot  = &Abuf[wv][lane][0];       // this lane's 32-K staging row
    const _Float16* arow0 = &Abuf[wv][l15][0];        // A rows, M-tile 0
    const _Float16* arow1 = &Abuf[wv][16 + l15][0];   // A rows, M-tile 1

    for (int tile = blockIdx.x; tile < nTiles; tile += gridDim.x) {
        const int row = tile * ROWS_PER_BLOCK + wv * 32 + lane;
        const float x0 = X[row * 3 + 0];
        const float x1 = X[row * 3 + 1];
        const float x2 = X[row * 3 + 2];

        // ================= layer 1: K=64 (2 chunks), N=32 (2 tiles) ========
        v8f a00 = {}, a01 = {}, a10 = {}, a11 = {};
        {   // chunk 0: features x0, x1
            kan_feature(slot, x0);
            kan_feature(slot + 16, x1);
            WAVE_DS_FENCE();
            v16h A0 = frag16(arow0 + hsA, arow0 + 16 + hsA);
            v16h A1 = frag16(arow1 + hsA, arow1 + 16 + hsA);
            const _Float16* c0 = &W1[l15 * 64];
            const _Float16* c1 = &W1[(16 + l15) * 64];
            v16h B0 = frag16(c0 + hsA, c0 + 16 + hsA);
            v16h B1 = frag16(c1 + hsA, c1 + 16 + hsA);
            a00 = wmma16(A0, B0, a00); a01 = wmma16(A0, B1, a01);
            a10 = wmma16(A1, B0, a10); a11 = wmma16(A1, B1, a11);
        }
        {   // chunk 1: feature x2 + zero padding feature
            kan_feature(slot, x2);
            v8h z = {};
            *(v8h*)(slot + 16) = z;
            *(v8h*)(slot + 24) = z;
            WAVE_DS_FENCE();
            v16h A0 = frag16(arow0 + hsA, arow0 + 16 + hsA);
            v16h A1 = frag16(arow1 + hsA, arow1 + 16 + hsA);
            const _Float16* c0 = &W1[l15 * 64] + 32;
            const _Float16* c1 = &W1[(16 + l15) * 64] + 32;
            v16h B0 = frag16(c0 + hsA, c0 + 16 + hsA);
            v16h B1 = frag16(c1 + hsA, c1 + 16 + hsA);
            a00 = wmma16(A0, B0, a00); a01 = wmma16(A0, B1, a01);
            a10 = wmma16(A1, B0, a10); a11 = wmma16(A1, B1, a11);
        }
        // transpose D tiles -> per-lane rows h1[0..31]
        #pragma unroll
        for (int v = 0; v < 8; ++v) {
            Hbuf[wv][mhi + v][l15]           = a00[v];
            Hbuf[wv][mhi + v][16 + l15]      = a01[v];
            Hbuf[wv][16 + mhi + v][l15]      = a10[v];
            Hbuf[wv][16 + mhi + v][16 + l15] = a11[v];
        }
        WAVE_DS_FENCE();
        float h1[32];
        #pragma unroll
        for (int j = 0; j < 32; ++j) h1[j] = Hbuf[wv][lane][j];

        // ================= layer 2: K=512 (16 chunks), N=16 ================
        v8f c20 = {}, c21 = {};
        #pragma unroll
        for (int c = 0; c < 16; ++c) {
            kan_feature(slot,      h1[2 * c]);
            kan_feature(slot + 16, h1[2 * c + 1]);
            WAVE_DS_FENCE();
            v16h A0 = frag16(arow0 + hsA, arow0 + 16 + hsA);
            v16h A1 = frag16(arow1 + hsA, arow1 + 16 + hsA);
            const _Float16* cb = &W2[l15 * 512] + c * 32;
            v16h B = frag16(cb + hsA, cb + 16 + hsA);
            c20 = wmma16(A0, B, c20);
            c21 = wmma16(A1, B, c21);
        }
        #pragma unroll
        for (int v = 0; v < 8; ++v) {
            Hbuf[wv][mhi + v][l15]      = c20[v];
            Hbuf[wv][16 + mhi + v][l15] = c21[v];
        }
        WAVE_DS_FENCE();
        float h2[16];
        #pragma unroll
        for (int j = 0; j < 16; ++j) h2[j] = Hbuf[wv][lane][j];

        // ================= layer 3: K=256 (8 chunks), N=1 (padded 16) ======
        v8f c30 = {}, c31 = {};
        #pragma unroll
        for (int c = 0; c < 8; ++c) {
            kan_feature(slot,      h2[2 * c]);
            kan_feature(slot + 16, h2[2 * c + 1]);
            WAVE_DS_FENCE();
            v16h A0 = frag16(arow0 + hsA, arow0 + 16 + hsA);
            v16h A1 = frag16(arow1 + hsA, arow1 + 16 + hsA);
            const _Float16* cb = &W3[l15 * 256] + c * 32;
            v16h B = frag16(cb + hsA, cb + 16 + hsA);
            c30 = wmma16(A0, B, c30);
            c31 = wmma16(A1, B, c31);
        }
        #pragma unroll
        for (int v = 0; v < 8; ++v) {
            Hbuf[wv][mhi + v][l15]      = c30[v];
            Hbuf[wv][16 + mhi + v][l15] = c31[v];
        }
        WAVE_DS_FENCE();
        Out[row] = Hbuf[wv][lane][0];   // column 0 = the single real output
    }
}

extern "C" void kernel_launch(void* const* d_in, const int* in_sizes, int n_in,
                              void* d_out, int out_size, void* d_ws, size_t ws_size,
                              hipStream_t stream) {
    (void)n_in; (void)d_ws; (void)ws_size; (void)out_size;
    const float* X   = (const float*)d_in[0];
    const float* bw1 = (const float*)d_in[1];
    const float* sw1 = (const float*)d_in[2];
    const float* bw2 = (const float*)d_in[3];
    const float* sw2 = (const float*)d_in[4];
    const float* bw3 = (const float*)d_in[5];
    const float* sw3 = (const float*)d_in[6];
    float* Out = (float*)d_out;

    const int B = in_sizes[0] / 3;                 // 262144
    const int nTiles = B / ROWS_PER_BLOCK;         // 2048 (exact)
    int blocks = nTiles < 1024 ? nTiles : 1024;

    kan3_wmma_kernel<<<blocks, TPB, 0, stream>>>(
        X, bw1, sw1, bw2, sw2, bw3, sw3, Out, nTiles);
}
